// MultiHeadedAttention_50087908606809
// MI455X (gfx1250) — matrix operand order repaired and checked by reading, not compiled, not run on hardware
//
#include <hip/hip_runtime.h>

// ---------------------------------------------------------------------------
// MultiHeadedAttention for MI455X (gfx1250, wave32, WMMA).
// B=8, L=1024, D=512, H=8, HD=64.
//   d_out = [ out (B,L,D) f32 | p (B,H,L,L) f32 ]
// Pipeline:
//   k0: cast inputs & weights f32 -> bf16 (workspace)
//   k1: per-(b,q) row sums Zr/Zt for rel- and time-softmax (head-independent!)
//   k2: projection GEMMs via v_wmma_f32_16x16x32_bf16; Q pre-scaled 1/sqrt(HD),
//       V written transposed (B,H,HD,L) so P@V B-fragments are contiguous.
//   k3: fused attention per 16-row q-tile (1 wave/block, 80KB dynamic LDS):
//       scores WMMA -> LDS (K-fragments double-buffered), masked softmax blend
//       using precomputed Zr/Zt, rel/ts rows async-staged to LDS
//       (global_load_async_to_lds_b128, guarded), p -> global+LDS, P@V -> out.
// ---------------------------------------------------------------------------

#define BQ  8
#define LQ  1024
#define DQ  512
#define HQ  8
#define HDQ 64

#if defined(__has_builtin)
#if __has_builtin(__builtin_amdgcn_global_load_async_to_lds_b128) && \
    __has_builtin(__builtin_amdgcn_s_wait_asynccnt)
#define USE_ASYNC_LDS 1
#endif
#endif

typedef __attribute__((ext_vector_type(16))) __bf16 v16bf;
typedef __attribute__((ext_vector_type(8)))  float  v8f;

union FragB16 {
    v16bf        v;
    uint4        u[2];
    unsigned int w[8];
};

__device__ __forceinline__ unsigned short f2bf(float f) {
    unsigned int u = __float_as_uint(f);
    unsigned int r = (u + 0x7FFFu + ((u >> 16) & 1u)) >> 16;   // RNE
    return (unsigned short)r;
}

// ---------------------------------------------------------------- cast kernel
__global__ void cast_f32_to_bf16(const float* __restrict__ src,
                                 unsigned short* __restrict__ dst, int n) {
    int i = blockIdx.x * blockDim.x + threadIdx.x;
    if (i < n) dst[i] = f2bf(src[i]);
}

// ----------------------------------------------- head-independent softmax sums
// Zr[n] = sum_{c unmasked} exp(rel[n,c]) ; Zt[n] = sum exp(exp(-|t[n,c]|)).
// (masked terms contribute exp(-1e5) ~ 0, matching the reference.)
// One wave per (b,q) row; 8 waves / block.
__global__ void __launch_bounds__(256)
relts_sums(const float* __restrict__ rel, const float* __restrict__ tsm,
           const unsigned char* __restrict__ mask,
           float* __restrict__ Zr, float* __restrict__ Zt)
{
    const int lane = threadIdx.x & 31;
    const int wave = threadIdx.x >> 5;
    const int n    = blockIdx.x * 8 + wave;        // row in (B*L)
    const size_t base = (size_t)n * LQ;
    const unsigned char* mrow = mask + base;
    const float* rrow = rel + base;
    const float* trow = tsm + base;

    float zr = 0.f, zt = 0.f;
    for (int c = lane; c < LQ; c += 32) {
        __builtin_prefetch(rrow + c + 256, 0, 1);
        __builtin_prefetch(trow + c + 256, 0, 1);
        if (!mrow[c]) {
            zr += __expf(rrow[c]);
            zt += __expf(__expf(-fabsf(trow[c])));
        }
    }
#pragma unroll
    for (int off = 16; off >= 1; off >>= 1) {
        zr += __shfl_xor(zr, off, 32);
        zt += __shfl_xor(zt, off, 32);
    }
    if (lane == 0) { Zr[n] = zr; Zt[n] = zt; }
}

// ------------------------------------------------------- projection GEMM (bf16)
// Y = X @ W^T + b.  X:(8192,512) bf16, W:(512,512) bf16 row-major.
// blockIdx.z selects {Q,K,V}.  8 waves/block; wave -> 16x64 output tile.
__global__ void __launch_bounds__(256)
proj_wmma(const unsigned short* __restrict__ Xq,
          const unsigned short* __restrict__ Xk,
          const unsigned short* __restrict__ Xv,
          const unsigned short* __restrict__ Wqb,
          const unsigned short* __restrict__ Wkb,
          const unsigned short* __restrict__ Wvb,
          const float* __restrict__ bq,
          const float* __restrict__ bk,
          const float* __restrict__ bv,
          unsigned short* __restrict__ Qb,   // (B,H,L,HD) bf16, scaled 1/sqrt(HD)
          unsigned short* __restrict__ Kb,   // (B,H,L,HD) bf16
          unsigned short* __restrict__ VT)   // (B,H,HD,L) bf16 (transposed)
{
    const int which = blockIdx.z;
    const unsigned short* X = (which == 0) ? Xq  : (which == 1) ? Xk  : Xv;
    const unsigned short* W = (which == 0) ? Wqb : (which == 1) ? Wkb : Wvb;
    const float* bias       = (which == 0) ? bq  : (which == 1) ? bk  : bv;
    const float scale       = (which == 0) ? 0.125f : 1.0f;   // 1/sqrt(64)

    const int lane = threadIdx.x & 31;
    const int wave = threadIdx.x >> 5;
    const int m    = lane & 15;     // row (A) / col (B,C) within 16-tile
    const int kh   = lane >> 4;     // K-half selector per WMMA lane layout

    const int row0 = blockIdx.x * 128 + wave * 16;
    const int col0 = blockIdx.y * 64;

    v8f acc[4];
    acc[0] = {}; acc[1] = {}; acc[2] = {}; acc[3] = {};

    const unsigned short* xrow = X + (size_t)(row0 + m) * DQ;

    for (int kb = 0; kb < DQ / 32; ++kb) {
        // A fragment 16x32: lanes 0-15 K{0..15}, lanes 16-31 K{16..31}
        FragB16 A;
        const unsigned short* xa = xrow + kb * 32 + kh * 16;
        A.u[0] = *(const uint4*)(xa);
        A.u[1] = *(const uint4*)(xa + 8);
        // Load all four B fragments first so WMMAs can use graduated waits.
        FragB16 Bf[4];
#pragma unroll
        for (int t = 0; t < 4; ++t) {
            const unsigned short* wr =
                W + (size_t)(col0 + t * 16 + m) * DQ + kb * 32 + kh * 16;
            Bf[t].u[0] = *(const uint4*)(wr);
            Bf[t].u[1] = *(const uint4*)(wr + 8);
        }
#pragma unroll
        for (int t = 0; t < 4; ++t)
            acc[t] = __builtin_amdgcn_wmma_f32_16x16x32_bf16(
                false, A.v, false, Bf[t].v, (short)0, acc[t], false, false);
    }

    // C layout: lane 0-15 -> N=lane, M=j ; lane 16-31 -> N=lane-16, M=j+8
#pragma unroll
    for (int t = 0; t < 4; ++t) {
        const int o  = col0 + t * 16 + m;           // output feature
        const int h  = o >> 6;
        const int hd = o & 63;
        const float bb = bias[o];
#pragma unroll
        for (int j = 0; j < 8; ++j) {
            const int n = row0 + j + 8 * kh;        // flat row in (B*L)
            const int b = n >> 10;
            const int l = n & 1023;
            const unsigned short bfv = f2bf((acc[t][j] + bb) * scale);
            if (which == 2)
                VT[(((size_t)(b * HQ + h)) * HDQ + hd) * LQ + l] = bfv;
            else if (which == 1)
                Kb[(((size_t)(b * HQ + h)) * LQ + l) * HDQ + hd] = bfv;
            else
                Qb[(((size_t)(b * HQ + h)) * LQ + l) * HDQ + hd] = bfv;
        }
    }
}

#ifdef USE_ASYNC_LDS
// Builtin prototype (from hipcc diagnostic): (v4i AS1*, v4i AS3*, imm, imm).
typedef int async_v4i __attribute__((vector_size(16)));
typedef async_v4i __attribute__((address_space(1)))* async_gp;
typedef async_v4i __attribute__((address_space(3)))* async_lp;

// Stage one (rel,ts) row pair (1024 f32 each) into LDS via async-to-LDS b128.
// 8 issues per array per wave (32 lanes x 16B = 512B per issue).
__device__ __forceinline__ void stage_row(const float* rrow, const float* trow,
                                          float* stg, int lane) {
#pragma unroll
    for (int i = 0; i < 8; ++i) {
        const int c = i * 128 + lane * 4;
        __builtin_amdgcn_global_load_async_to_lds_b128(
            (async_gp)(rrow + c), (async_lp)(stg + c), 0, 0);
        __builtin_amdgcn_global_load_async_to_lds_b128(
            (async_gp)(trow + c), (async_lp)(stg + 1024 + c), 0, 0);
    }
}
#endif

// ----------------------------------------------------------- fused attention
// One wave per block; block handles one (b,h,q-tile of 16 rows).
// Dynamic LDS: 16*1024 f32 scores/p (64KB) + 2*2048 f32 rel/ts staging (16KB).
__global__ void __launch_bounds__(32)
attn_wmma(const unsigned short* __restrict__ Qb,
          const unsigned short* __restrict__ Kb,
          const unsigned short* __restrict__ VT,
          const float* __restrict__ rel,          // (B,L,L)
          const float* __restrict__ tsm,          // (B,L,L)
          const unsigned char* __restrict__ mask, // (B,L,L) bool
          const float* __restrict__ Zr,           // (B*L)
          const float* __restrict__ Zt,           // (B*L)
          const float* __restrict__ l1p,
          const float* __restrict__ l2p,
          float* __restrict__ outp,               // (B,L,D)
          float* __restrict__ pout)               // (B,H,L,L)
{
    extern __shared__ float sm[];                 // [16][1024] + staging
    float* stg = sm + 16 * LQ;                    // [2][2048]
    const int lane  = threadIdx.x & 31;
    const int m     = lane & 15;
    const int kh    = lane >> 4;
    const int bh    = blockIdx.x >> 6;            // b*H + h
    const int qt    = blockIdx.x & 63;
    const int b     = bh >> 3;
    const int h     = bh & 7;
    const int qbase = qt * 16;

    const float l1 = *l1p, l2 = *l2p;
    const float c1 = (1.f - l1) * (1.f - l2);
    const float c2 = l1 * (1.f - l2);
    const float c3 = l2;

    // ---------------- Stage 1: scores = (Q/sqrt(HD)) @ K^T  (HD=64 -> 2 WMMA)
    FragB16 A0, A1;
    {
        const unsigned short* qrow = Qb + ((size_t)bh * LQ + qbase + m) * HDQ;
        A0.u[0] = *(const uint4*)(qrow + kh * 16);
        A0.u[1] = *(const uint4*)(qrow + kh * 16 + 8);
        A1.u[0] = *(const uint4*)(qrow + 32 + kh * 16);
        A1.u[1] = *(const uint4*)(qrow + 32 + kh * 16 + 8);
    }
    {
        const unsigned short* kbase_p = Kb + (size_t)bh * LQ * HDQ;
        FragB16 X0, X1, Y0, Y1;   // double-buffered K-tile fragments
        {
            const unsigned short* krow = kbase_p + (size_t)m * HDQ;
            X0.u[0] = *(const uint4*)(krow + kh * 16);
            X0.u[1] = *(const uint4*)(krow + kh * 16 + 8);
            X1.u[0] = *(const uint4*)(krow + 32 + kh * 16);
            X1.u[1] = *(const uint4*)(krow + 32 + kh * 16 + 8);
        }
        for (int kb = 0; kb < LQ / 16; kb += 2) {
            {   // prefetch kb+1 tile
                const unsigned short* krow =
                    kbase_p + (size_t)((kb + 1) * 16 + m) * HDQ;
                Y0.u[0] = *(const uint4*)(krow + kh * 16);
                Y0.u[1] = *(const uint4*)(krow + kh * 16 + 8);
                Y1.u[0] = *(const uint4*)(krow + 32 + kh * 16);
                Y1.u[1] = *(const uint4*)(krow + 32 + kh * 16 + 8);
            }
            v8f acc = {};
            acc = __builtin_amdgcn_wmma_f32_16x16x32_bf16(
                false, A0.v, false, X0.v, (short)0, acc, false, false);
            acc = __builtin_amdgcn_wmma_f32_16x16x32_bf16(
                false, A1.v, false, X1.v, (short)0, acc, false, false);
#pragma unroll
            for (int j = 0; j < 8; ++j)
                sm[(j + 8 * kh) * LQ + kb * 16 + m] = acc[j];

            if (kb + 2 < LQ / 16) {   // prefetch kb+2 tile
                const unsigned short* krow =
                    kbase_p + (size_t)((kb + 2) * 16 + m) * HDQ;
                X0.u[0] = *(const uint4*)(krow + kh * 16);
                X0.u[1] = *(const uint4*)(krow + kh * 16 + 8);
                X1.u[0] = *(const uint4*)(krow + 32 + kh * 16);
                X1.u[1] = *(const uint4*)(krow + 32 + kh * 16 + 8);
            }
            v8f acc2 = {};
            acc2 = __builtin_amdgcn_wmma_f32_16x16x32_bf16(
                false, A0.v, false, Y0.v, (short)0, acc2, false, false);
            acc2 = __builtin_amdgcn_wmma_f32_16x16x32_bf16(
                false, A1.v, false, Y1.v, (short)0, acc2, false, false);
#pragma unroll
            for (int j = 0; j < 8; ++j)
                sm[(j + 8 * kh) * LQ + (kb + 1) * 16 + m] = acc2[j];
        }
    }
    __syncthreads();

#ifdef USE_ASYNC_LDS
    stage_row(rel + ((size_t)b * LQ + qbase) * LQ,
              tsm + ((size_t)b * LQ + qbase) * LQ, stg, lane);
#endif

    // ---------------- Stage 2+3: softmax of scores (max+sum from LDS), blend
    //                  with precomputed Zr/Zt, write p (global + LDS).
    for (int r = 0; r < 16; ++r) {
        const int gq = qbase + r;
        float* srow = sm + r * LQ;
        const unsigned char* mrow = mask + ((size_t)b * LQ + gq) * LQ;
        const float* rrow = rel + ((size_t)b * LQ + gq) * LQ;
        const float* trow = tsm + ((size_t)b * LQ + gq) * LQ;

#ifdef USE_ASYNC_LDS
        if (r < 15)   // stage next row while we reduce this one
            stage_row(rrow + LQ, trow + LQ, stg + ((r + 1) & 1) * 2048, lane);
#endif

        float smax = -3.0e38f;
        for (int c = lane; c < LQ; c += 32)
            if (!mrow[c]) smax = fmaxf(smax, srow[c]);
#pragma unroll
        for (int off = 16; off >= 1; off >>= 1)
            smax = fmaxf(smax, __shfl_xor(smax, off, 32));

        float zs = 0.f;
        for (int c = lane; c < LQ; c += 32)
            if (!mrow[c]) zs += __expf(srow[c] - smax);
#pragma unroll
        for (int off = 16; off >= 1; off >>= 1)
            zs += __shfl_xor(zs, off, 32);

        const float is = c1 / zs;
        const float ir = c2 / Zr[b * LQ + gq];
        const float it = c3 / Zt[b * LQ + gq];

#ifdef USE_ASYNC_LDS
        if (r < 15) __builtin_amdgcn_s_wait_asynccnt(16);  // this row's stage done
        else        __builtin_amdgcn_s_wait_asynccnt(0);
        const float* stgc = stg + (r & 1) * 2048;
#endif
        float* prow = pout + ((size_t)bh * LQ + gq) * LQ;
        for (int c = lane; c < LQ; c += 32) {
#ifdef USE_ASYNC_LDS
            const float rv = stgc[c];
            const float tv = stgc[1024 + c];
#else
            __builtin_prefetch(rrow + c + 256, 0, 1);
            __builtin_prefetch(trow + c + 256, 0, 1);
            const float rv = rrow[c];
            const float tv = trow[c];
#endif
            float pv = 0.f;
            if (gq != 0 && !mrow[c]) {      // row 0 of each (b,h) is zeroed
                pv = is * __expf(srow[c] - smax)
                   + ir * __expf(rv)
                   + it * __expf(__expf(-fabsf(tv)));
            }
            srow[c] = pv;                   // p into LDS for the P@V stage
            prow[c] = pv;                   // p to global output
        }
    }
    __syncthreads();

    // ---------------- Stage 4: out = P @ V   (K=1024 -> 32 WMMA per 16x16 tile)
    v8f oacc[4];
    oacc[0] = {}; oacc[1] = {}; oacc[2] = {}; oacc[3] = {};
    for (int kb = 0; kb < LQ / 32; ++kb) {
        FragB16 Bf[4];
#pragma unroll
        for (int t = 0; t < 4; ++t) {
            const unsigned short* vrow =
                VT + ((size_t)bh * HDQ + t * 16 + m) * LQ + kb * 32 + kh * 16;
            Bf[t].u[0] = *(const uint4*)(vrow);
            Bf[t].u[1] = *(const uint4*)(vrow + 8);
        }
        FragB16 A;
        const float* pr = sm + m * LQ + kb * 32 + kh * 16;
#pragma unroll
        for (int d = 0; d < 4; ++d) {
            unsigned int lo = f2bf(pr[2 * d]);
            unsigned int hi = f2bf(pr[2 * d + 1]);
            A.w[d] = lo | (hi << 16);
            lo = f2bf(pr[8 + 2 * d]);
            hi = f2bf(pr[8 + 2 * d + 1]);
            A.w[4 + d] = lo | (hi << 16);
        }
#pragma unroll
        for (int t = 0; t < 4; ++t)
            oacc[t] = __builtin_amdgcn_wmma_f32_16x16x32_bf16(
                false, A.v, false, Bf[t].v, (short)0, oacc[t], false, false);
    }
#pragma unroll
    for (int t = 0; t < 4; ++t) {
        const int d = h * HDQ + t * 16 + m;
#pragma unroll
        for (int j = 0; j < 8; ++j) {
            const int l = qbase + j + 8 * kh;
            outp[((size_t)b * LQ + l) * DQ + d] = oacc[t][j];
        }
    }
}

// ---------------------------------------------------------------------------
extern "C" void kernel_launch(void* const* d_in, const int* in_sizes, int n_in,
                              void* d_out, int out_size, void* d_ws, size_t ws_size,
                              hipStream_t stream) {
    (void)in_sizes; (void)n_in; (void)out_size; (void)ws_size;

    const float* query = (const float*)d_in[0];
    const float* key   = (const float*)d_in[1];
    const float* value = (const float*)d_in[2];
    const float* rel   = (const float*)d_in[3];
    const float* l1    = (const float*)d_in[4];
    const float* l2    = (const float*)d_in[5];
    const float* tsm   = (const float*)d_in[6];
    const unsigned char* mask = (const unsigned char*)d_in[7];
    const float* Wq = (const float*)d_in[8];
    const float* bq = (const float*)d_in[9];
    const float* Wk = (const float*)d_in[10];
    const float* bk = (const float*)d_in[11];
    const float* Wv = (const float*)d_in[12];
    const float* bv = (const float*)d_in[13];

    // Workspace carve-out (~49.6 MB total).
    const size_t NX = (size_t)BQ * LQ * DQ;   // 4,194,304 elements
    const size_t NW = (size_t)DQ * DQ;        //   262,144 elements
    const size_t NR = (size_t)BQ * LQ;        //     8,192 rows
    char* ws = (char*)d_ws;
    unsigned short* Xq  = (unsigned short*)ws; ws += NX * 2;
    unsigned short* Xk  = (unsigned short*)ws; ws += NX * 2;
    unsigned short* Xv  = (unsigned short*)ws; ws += NX * 2;
    unsigned short* Wqb = (unsigned short*)ws; ws += NW * 2;
    unsigned short* Wkb = (unsigned short*)ws; ws += NW * 2;
    unsigned short* Wvb = (unsigned short*)ws; ws += NW * 2;
    unsigned short* Qb  = (unsigned short*)ws; ws += NX * 2;  // (B,H,L,HD)
    unsigned short* Kb  = (unsigned short*)ws; ws += NX * 2;  // (B,H,L,HD)
    unsigned short* VT  = (unsigned short*)ws; ws += NX * 2;  // (B,H,HD,L)
    float*          Zr  = (float*)ws;          ws += NR * 4;
    float*          Zt  = (float*)ws;          ws += NR * 4;

    // Stage 0: f32 -> bf16 casts.
    cast_f32_to_bf16<<<(int)((NX + 255) / 256), 256, 0, stream>>>(query, Xq, (int)NX);
    cast_f32_to_bf16<<<(int)((NX + 255) / 256), 256, 0, stream>>>(key,   Xk, (int)NX);
    cast_f32_to_bf16<<<(int)((NX + 255) / 256), 256, 0, stream>>>(value, Xv, (int)NX);
    cast_f32_to_bf16<<<(int)((NW + 255) / 256), 256, 0, stream>>>(Wq, Wqb, (int)NW);
    cast_f32_to_bf16<<<(int)((NW + 255) / 256), 256, 0, stream>>>(Wk, Wkb, (int)NW);
    cast_f32_to_bf16<<<(int)((NW + 255) / 256), 256, 0, stream>>>(Wv, Wvb, (int)NW);

    // Stage 1: head-independent rel/time softmax normalizers (B*L rows).
    relts_sums<<<(int)(NR / 8), 256, 0, stream>>>(rel, tsm, mask, Zr, Zt);

    // Stage 2: projection GEMMs (z picks Q/K/V).
    dim3 pgrid((BQ * LQ) / 128, DQ / 64, 3);   // (64, 8, 3)
    proj_wmma<<<pgrid, 256, 0, stream>>>(Xq, Xk, Xv, Wqb, Wkb, Wvb,
                                         bq, bk, bv, Qb, Kb, VT);

    // Stage 3: fused attention.  d_out = [ out (B,L,D) | p (B,H,L,L) ].
    float* outp = (float*)d_out;
    float* pout = outp + (size_t)BQ * LQ * DQ;
    const size_t shmem = (size_t)(16 * LQ + 2 * 2048) * sizeof(float);  // 80KB
    attn_wmma<<<BQ * HQ * (LQ / 16), 32, shmem, stream>>>(
        Qb, Kb, VT, rel, tsm, mask, Zr, Zt, l1, l2, outp, pout);
}
